// QwenMoeTransformerDecoder_10685878632760
// MI455X (gfx1250) — compile-verified, matrix operand order checked by reading, not run on hardware
//
#include <hip/hip_runtime.h>
#include <cstdint>
#include <cstddef>

// ---------------------------------------------------------------------------
// QwenMoE decoder block for MI455X (gfx1250, wave32, WMMA).
// Compute-bound (~0.71 TFLOP vs ~1GB HBM traffic @ 23.3TB/s): all big GEMMs
// run v_wmma_f32_16x16x32_bf16. Weights are transposed+converted to bf16 once
// (so weight tiles stage into LDS as raw b128 copies via the gfx1250 async
// global->LDS path), activations converted once.
// ---------------------------------------------------------------------------

#define T_TOK 2048
#define H_DIM 2048
#define E_EXP 16
#define I_DIM 1408
#define IS_DIM 5632

typedef __attribute__((ext_vector_type(16))) __bf16        v16bf;
typedef __attribute__((ext_vector_type(8)))  float         v8f;
typedef __attribute__((ext_vector_type(4)))  unsigned int  v4u;
typedef __attribute__((ext_vector_type(4)))  int           v4i;

union FragAB { v16bf v; v4u u[2]; };

#if __has_builtin(__builtin_amdgcn_global_load_async_to_lds_b128) && \
    __has_builtin(__builtin_amdgcn_s_wait_asynccnt)
#define USE_ASYNC_LDS 1
#else
#define USE_ASYNC_LDS 0
#endif

// ------------------------------ small kernels ------------------------------

__global__ __launch_bounds__(256) void zero_f32(float* __restrict__ p, int n) {
  int i = blockIdx.x * 256 + threadIdx.x;
  if (i < n) p[i] = 0.f;
}

__global__ __launch_bounds__(256) void cvt_to_bf16(const float* __restrict__ x,
                                                   __bf16* __restrict__ y, int n) {
  int i = blockIdx.x * 256 + threadIdx.x;
  if (i < n) y[i] = (__bf16)x[i];
}

// Tiled transpose + fp32->bf16 convert:  outT[n*K + k] = (bf16) in[k*ldin + n]
// grid = (N/32, K/32), block = 256 (32x8).
__global__ __launch_bounds__(256) void transpose_cvt(const float* __restrict__ in,
                                                     __bf16* __restrict__ outT,
                                                     int K, int ldin) {
  __shared__ float tile[32][33];
  const int kb = blockIdx.y * 32, nb = blockIdx.x * 32;
  const int tx = threadIdx.x & 31, ty = threadIdx.x >> 5;
#pragma unroll
  for (int i = 0; i < 32; i += 8)
    tile[ty + i][tx] = in[(size_t)(kb + ty + i) * ldin + nb + tx];
  __syncthreads();
#pragma unroll
  for (int i = 0; i < 32; i += 8)
    outT[(size_t)(nb + ty + i) * K + kb + tx] = (__bf16)tile[tx][ty + i];
}

// Router: logits = x @ rw [T,16]; softmax -> top-2 -> renorm -> dense [E,T].
__global__ __launch_bounds__(128) void router_kernel(const float* __restrict__ x,
                                                     const float* __restrict__ rw,
                                                     float* __restrict__ logits_out,
                                                     float* __restrict__ routew) {
  int t = blockIdx.x * 128 + threadIdx.x;
  if (t >= T_TOK) return;
  float acc[E_EXP];
#pragma unroll
  for (int e = 0; e < E_EXP; ++e) acc[e] = 0.f;
  const float* xr = x + (size_t)t * H_DIM;
  for (int h = 0; h < H_DIM; ++h) {
    float xv = xr[h];
    const float* w = rw + (size_t)h * E_EXP;
#pragma unroll
    for (int e = 0; e < E_EXP; ++e) acc[e] += xv * w[e];
  }
#pragma unroll
  for (int e = 0; e < E_EXP; ++e) logits_out[(size_t)t * E_EXP + e] = acc[e];
  float mx = acc[0];
#pragma unroll
  for (int e = 1; e < E_EXP; ++e) mx = fmaxf(mx, acc[e]);
  float p[E_EXP];
#pragma unroll
  for (int e = 0; e < E_EXP; ++e) p[e] = __expf(acc[e] - mx);
  int i1 = 0; float p1 = p[0];
#pragma unroll
  for (int e = 1; e < E_EXP; ++e) if (p[e] > p1) { p1 = p[e]; i1 = e; }
  int i2 = (i1 == 0) ? 1 : 0; float p2 = p[i2];
#pragma unroll
  for (int e = 0; e < E_EXP; ++e)
    if (e != i1 && p[e] > p2) { p2 = p[e]; i2 = e; }
  float inv = 1.f / (p1 + p2);
  float w1 = p1 * inv, w2 = p2 * inv;
#pragma unroll
  for (int e = 0; e < E_EXP; ++e)
    routew[(size_t)e * T_TOK + t] = (e == i1) ? w1 : ((e == i2) ? w2 : 0.f);
}

// sgv[t] = sigmoid(x[t,:] @ shared_gate_w[:,0])
__global__ __launch_bounds__(128) void gatevec_kernel(const float* __restrict__ x,
                                                      const float* __restrict__ gw,
                                                      float* __restrict__ sgv) {
  int t = blockIdx.x * 128 + threadIdx.x;
  if (t >= T_TOK) return;
  const float* xr = x + (size_t)t * H_DIM;
  float d = 0.f;
  for (int h = 0; h < H_DIM; ++h) d += xr[h] * gw[h];
  sgv[t] = 1.f / (1.f + __expf(-d));
}

// ------------------------------ WMMA GEMM ----------------------------------
// C-tile 128x128 per block (256 threads = 8 waves, wave tile 32x64).
// A  : bf16 row-major [M,K]   (activations)
// Bt : bf16 row-major [N,K]   (pre-transposed weights)
// MODE 0: C    = silu(acc)            (gate matmul, fp32)
// MODE 1: Cb   = bf16(acc * gmul)     (up matmul * silu(gate))
// MODE 2: C   += rowscale[m] * acc    (down matmul, routed/gated accumulate)

#define TM 128
#define TN 128
#define KSTEP 32
#define LROW 20  // dwords per LDS row: 32 bf16 + pad; 16B aligned, bank-friendly

template <int MODE>
__global__ __launch_bounds__(256) void gemm_wmma(const __bf16* __restrict__ A,
                                                 const __bf16* __restrict__ Bt,
                                                 float* __restrict__ C,
                                                 __bf16* __restrict__ Cb,
                                                 const float* __restrict__ gmul,
                                                 const float* __restrict__ rowscale,
                                                 int M, int N, int K) {
  __shared__ unsigned int sA[2][TM * LROW];
  __shared__ unsigned int sB[2][TN * LROW];

  const int tid  = threadIdx.x;
  const int lane = tid & 31;
  const int wave = tid >> 5;
  const int l15  = lane & 15;
  const int half = lane >> 4;
  const int m0   = blockIdx.y * TM;
  const int n0   = blockIdx.x * TN;
  const int mtb  = (wave >> 1) * 32;  // wave's M sub-tile (0..96)
  const int ntb  = (wave & 1) * 64;   // wave's N sub-tile (0/64)

  v8f acc[8];
#pragma unroll
  for (int i = 0; i < 8; ++i)
    acc[i] = v8f{0.f, 0.f, 0.f, 0.f, 0.f, 0.f, 0.f, 0.f};

  // Staging: 2 threads per row, each copies 32B (2 x b128).
  const int ar   = tid >> 1;
  const int aseg = tid & 1;
  const __bf16* gA = A + (size_t)(m0 + ar) * K + aseg * 16;
  const __bf16* gB = Bt + (size_t)(n0 + ar) * K + aseg * 16;
  const int ld = ar * LROW + aseg * 8;

  // 16-bit A/B fragment gather (ISA 7.12.2): lane = row, half-wave selects the
  // K sub-block, pairs K-contiguous -> 2 x ds_load_b128 per fragment.
  auto compute = [&](int cur) {
    FragAB afr[2], bfr[4];
#pragma unroll
    for (int i = 0; i < 2; ++i) {
      const unsigned int* ap = &sA[cur][(mtb + i * 16 + l15) * LROW + half * 4];
      afr[i].u[0] = *(const v4u*)ap;
      afr[i].u[1] = *(const v4u*)(ap + 8);
    }
#pragma unroll
    for (int j = 0; j < 4; ++j) {
      const unsigned int* bp = &sB[cur][(ntb + j * 16 + l15) * LROW + half * 4];
      bfr[j].u[0] = *(const v4u*)bp;
      bfr[j].u[1] = *(const v4u*)(bp + 8);
    }
#pragma unroll
    for (int i = 0; i < 2; ++i)
#pragma unroll
      for (int j = 0; j < 4; ++j)
        acc[i * 4 + j] = __builtin_amdgcn_wmma_f32_16x16x32_bf16(
            false, afr[i].v, false, bfr[j].v, (short)0, acc[i * 4 + j], false,
            false);
  };

  const int LT = K / KSTEP;

#if USE_ASYNC_LDS
  // gfx1250 async global->LDS DMA, double-buffered; tracked by ASYNCcnt.
  // Builtin signature (per hipcc diagnostic): (v4i AS1*, v4i AS3*, Imm, Imm).
#define ASYNC_CP32(gp, lp)                                                     \
  do {                                                                         \
    __builtin_amdgcn_global_load_async_to_lds_b128(                            \
        (__attribute__((address_space(1))) v4i*)(gp),                          \
        (__attribute__((address_space(3))) v4i*)(lp), 0, 0);                   \
    __builtin_amdgcn_global_load_async_to_lds_b128(                            \
        (__attribute__((address_space(1))) v4i*)(gp),                          \
        (__attribute__((address_space(3))) v4i*)(lp), 16, 0);                  \
  } while (0)

  ASYNC_CP32(gA, &sA[0][ld]);
  ASYNC_CP32(gB, &sB[0][ld]);
  for (int it = 0; it < LT; ++it) {
    const int cur = it & 1;
    if (it + 1 < LT) {  // prefetch next tile into the other buffer
      ASYNC_CP32(gA + (size_t)(it + 1) * KSTEP, &sA[1 - cur][ld]);
      ASYNC_CP32(gB + (size_t)(it + 1) * KSTEP, &sB[1 - cur][ld]);
      __builtin_amdgcn_s_wait_asynccnt(4);  // current tile's 4 copies done
    } else {
      __builtin_amdgcn_s_wait_asynccnt(0);
    }
    __syncthreads();
    compute(cur);
    __syncthreads();
  }
#else
  // Fallback: synchronous b128 loads + LDS stores, double buffered
  // (one barrier per K-step: writes always target the buffer not being read).
  for (int it = 0; it < LT; ++it) {
    const int cur = it & 1;
    const size_t k0 = (size_t)it * KSTEP;
    v4u a0 = *(const v4u*)(gA + k0);
    v4u a1 = *(const v4u*)(gA + k0 + 8);
    v4u b0 = *(const v4u*)(gB + k0);
    v4u b1 = *(const v4u*)(gB + k0 + 8);
    *(v4u*)&sA[cur][ld]     = a0;
    *(v4u*)&sA[cur][ld + 4] = a1;
    *(v4u*)&sB[cur][ld]     = b0;
    *(v4u*)&sB[cur][ld + 4] = b1;
    __syncthreads();
    compute(cur);
  }
#endif

  // ---- epilogue ----  C/D layout: VGPR r -> row (half*8 + r), lane15 -> col.
  const int mbase = m0 + mtb + half * 8;
#pragma unroll
  for (int i = 0; i < 2; ++i) {
#pragma unroll
    for (int j = 0; j < 4; ++j) {
      const int col = n0 + ntb + j * 16 + l15;
#pragma unroll
      for (int r = 0; r < 8; ++r) {
        const int row = mbase + i * 16 + r;
        float v = acc[i * 4 + j][r];
        size_t idx = (size_t)row * N + col;
        if (MODE == 0) {
          C[idx] = v / (1.f + __expf(-v));      // silu(gate)
        } else if (MODE == 1) {
          Cb[idx] = (__bf16)(v * gmul[idx]);    // up * silu(gate) -> bf16
        } else {
          C[idx] += rowscale[row] * v;          // routed / sigmoid-gated add
        }
      }
    }
  }
  (void)M;
}

// ------------------------------ host launch --------------------------------

extern "C" void kernel_launch(void* const* d_in, const int* in_sizes, int n_in,
                              void* d_out, int out_size, void* d_ws, size_t ws_size,
                              hipStream_t stream) {
  (void)in_sizes; (void)n_in; (void)out_size; (void)ws_size;

  const float* x      = (const float*)d_in[0];  // [T,H]
  const float* rw     = (const float*)d_in[1];  // [H,E]
  const float* egu    = (const float*)d_in[2];  // [E,H,2I]
  const float* eo     = (const float*)d_in[3];  // [E,I,H]
  const float* sgw    = (const float*)d_in[4];  // [H,IS]
  const float* siw    = (const float*)d_in[5];  // [H,IS]
  const float* sow    = (const float*)d_in[6];  // [IS,H]
  const float* sgatew = (const float*)d_in[7];  // [H,1]

  float* out    = (float*)d_out;                // [T,H]
  float* logits = out + (size_t)T_TOK * H_DIM;  // [T,E]

  // workspace carve-out (~101 MB)
  size_t off = 0;
  char* base = (char*)d_ws;
  auto carve = [&](size_t bytes) -> char* {
    char* p = base + off;
    off += (bytes + 255) & ~(size_t)255;
    return p;
  };
  __bf16* xbf    = (__bf16*)carve((size_t)T_TOK * H_DIM * 2);
  float*  routew = (float*)carve((size_t)E_EXP * T_TOK * 4);
  float*  sgv    = (float*)carve((size_t)T_TOK * 4);
  float*  gbuf   = (float*)carve((size_t)T_TOK * IS_DIM * 4);    // silu(gate)
  __bf16* hbf    = (__bf16*)carve((size_t)T_TOK * IS_DIM * 2);   // swiglu acts
  __bf16* btw    = (__bf16*)carve((size_t)IS_DIM * H_DIM * 2);   // W^T bf16

  const dim3 blk(256);
  const int TH = T_TOK * H_DIM;

  zero_f32<<<(TH + 255) / 256, blk, 0, stream>>>(out, TH);
  cvt_to_bf16<<<(TH + 255) / 256, blk, 0, stream>>>(x, xbf, TH);
  router_kernel<<<(T_TOK + 127) / 128, 128, 0, stream>>>(x, rw, logits, routew);
  gatevec_kernel<<<(T_TOK + 127) / 128, 128, 0, stream>>>(x, sgatew, sgv);

  // ---- shared expert: swiglu MLP, sigmoid-gated accumulate into out ----
  transpose_cvt<<<dim3(IS_DIM / 32, H_DIM / 32), blk, 0, stream>>>(sgw, btw, H_DIM, IS_DIM);
  gemm_wmma<0><<<dim3(IS_DIM / TN, T_TOK / TM), blk, 0, stream>>>(
      xbf, btw, gbuf, nullptr, nullptr, nullptr, T_TOK, IS_DIM, H_DIM);
  transpose_cvt<<<dim3(IS_DIM / 32, H_DIM / 32), blk, 0, stream>>>(siw, btw, H_DIM, IS_DIM);
  gemm_wmma<1><<<dim3(IS_DIM / TN, T_TOK / TM), blk, 0, stream>>>(
      xbf, btw, nullptr, hbf, gbuf, nullptr, T_TOK, IS_DIM, H_DIM);
  transpose_cvt<<<dim3(H_DIM / 32, IS_DIM / 32), blk, 0, stream>>>(sow, btw, IS_DIM, H_DIM);
  gemm_wmma<2><<<dim3(H_DIM / TN, T_TOK / TM), blk, 0, stream>>>(
      hbf, btw, out, nullptr, nullptr, sgv, T_TOK, H_DIM, IS_DIM);

  // ---- 16 experts, dense over all tokens, routed accumulate ----
  for (int e = 0; e < E_EXP; ++e) {
    const float* Bgu = egu + (size_t)e * H_DIM * (2 * I_DIM);
    const float* Bo  = eo + (size_t)e * I_DIM * H_DIM;
    transpose_cvt<<<dim3(I_DIM / 32, H_DIM / 32), blk, 0, stream>>>(Bgu, btw, H_DIM, 2 * I_DIM);
    gemm_wmma<0><<<dim3(I_DIM / TN, T_TOK / TM), blk, 0, stream>>>(
        xbf, btw, gbuf, nullptr, nullptr, nullptr, T_TOK, I_DIM, H_DIM);
    transpose_cvt<<<dim3(I_DIM / 32, H_DIM / 32), blk, 0, stream>>>(Bgu + I_DIM, btw, H_DIM, 2 * I_DIM);
    gemm_wmma<1><<<dim3(I_DIM / TN, T_TOK / TM), blk, 0, stream>>>(
        xbf, btw, nullptr, hbf, gbuf, nullptr, T_TOK, I_DIM, H_DIM);
    transpose_cvt<<<dim3(H_DIM / 32, I_DIM / 32), blk, 0, stream>>>(Bo, btw, I_DIM, H_DIM);
    gemm_wmma<2><<<dim3(H_DIM / TN, T_TOK / TM), blk, 0, stream>>>(
        hbf, btw, out, nullptr, nullptr, routew + (size_t)e * T_TOK, T_TOK, H_DIM, I_DIM);
  }
}
